// AttnGRU_8924942041746
// MI455X (gfx1250) — matrix operand order repaired and checked by reading, not compiled, not run on hardware
//
#include <hip/hip_runtime.h>
#include <hip/hip_bf16.h>
#include <math.h>

typedef __bf16 bf16;
typedef __attribute__((ext_vector_type(16))) __bf16 v16bf;
typedef __attribute__((ext_vector_type(8)))  __bf16 v8bf;
typedef __attribute__((ext_vector_type(8)))  float  v8f;

#define B_DIM 512
#define S_DIM 128
#define H_DIM 512

// LDS row stride padding: 520 bf16 -> 260 dwords; 260 % 64 == 4, so the 16
// A-fragment lanes land on banks 4*lane (conflict-free b128 fragment loads).
#define LDSB_STRIDE (H_DIM + 8)

#define WMMA_BF16(a, b, c) \
    __builtin_amdgcn_wmma_f32_16x16x32_bf16(false, (a), false, (b), (short)0, (c), false, false)

// ---------------------------------------------------------------------------
// Convert the four H x H weight matrices to bf16 (row-major, o-major) once.
// ---------------------------------------------------------------------------
__global__ void cvt_weights(const float* __restrict__ wr, const float* __restrict__ ur,
                            const float* __restrict__ w,  const float* __restrict__ u,
                            bf16* __restrict__ owr, bf16* __restrict__ our,
                            bf16* __restrict__ ow,  bf16* __restrict__ ou) {
    int i = blockIdx.x * blockDim.x + threadIdx.x;
    if (i < H_DIM * H_DIM) {
        owr[i] = (bf16)wr[i];
        our[i] = (bf16)ur[i];
        ow[i]  = (bf16)w[i];
        ou[i]  = (bf16)u[i];
    }
}

// ---------------------------------------------------------------------------
// Phase 1: fWr = facts @ Wr^T + Wr_b ; fW = facts @ W^T + W_b
// One block = one 16-row stripe of facts x all 512 output columns of BOTH
// matrices. The 16x512 f32 A tile is staged once into LDS as bf16 (facts read
// exactly once from HBM), then 8 waves run 4 N-tiles x 2 matrices = 8 WMMA
// chains each off the shared A fragment. All 8 B fragments of a K-chunk are
// loaded into live registers first (single load clause), then the 8-WMMA
// chain runs, so the chunk's L2 latency is exposed once, not per fragment.
// Output stored transposed to (S,B,H) with bias fused.
// ---------------------------------------------------------------------------
__global__ __launch_bounds__(256) void fact_gemm(
    const float* __restrict__ facts,
    const bf16*  __restrict__ wr16, const bf16* __restrict__ w16,
    const float* __restrict__ wr_bias, const float* __restrict__ w_bias,
    float* __restrict__ fwr_out, float* __restrict__ fw_out)
{
    __shared__ bf16 aT[16 * LDSB_STRIDE];

    const int lane = threadIdx.x & 31;
    const int wave = threadIdx.x >> 5;      // 0..7
    const int hi = lane >> 4;               // K-half select (this code's contiguous k = 16*hi + s)
    const int lo = lane & 15;
    const int rowBase = blockIdx.x * 16;    // facts flat rows (b*S + s)

    // Stage A: 16 rows x 512 f32 -> bf16 LDS. 16 threads/row, 32 elems each.
    {
        int r = threadIdx.x >> 4;
        int cBase = (threadIdx.x & 15) * 32;
        const float* src = facts + (size_t)(rowBase + r) * H_DIM + cBase;
        bf16* dst = &aT[r * LDSB_STRIDE + cBase];
#pragma unroll
        for (int j = 0; j < 32; ++j) dst[j] = (bf16)src[j];
    }
    __syncthreads();

    // This wave's 64 output columns: tiles c0/16 .. c0/16+3, for Wr and W.
    const int c0 = wave * 64;
    const bf16* bp[8];
#pragma unroll
    for (int tl = 0; tl < 4; ++tl) {
        bp[tl]     = wr16 + (size_t)(c0 + tl * 16 + lo) * H_DIM;
        bp[4 + tl] = w16  + (size_t)(c0 + tl * 16 + lo) * H_DIM;
    }

    v8f acc[8] = {};
    for (int kc = 0; kc < H_DIM; kc += 32) {
        // Phase a: issue ALL fragment loads for this chunk (A from LDS,
        // 8 x 32B of B from L2) while nothing waits on them yet.
        union { v16bf v; v8bf h[2]; } a;
        const bf16* ar = &aT[lo * LDSB_STRIDE + kc + hi * 16];
        a.h[0] = *(const v8bf*)(ar);
        a.h[1] = *(const v8bf*)(ar + 8);

        v16bf bfrag[8];
#pragma unroll
        for (int j = 0; j < 8; ++j)
            bfrag[j] = *(const v16bf*)(bp[j] + kc + hi * 16);

        // Phase b: 8-WMMA chain; waits on bfrag[j] are progressive.
#pragma unroll
        for (int j = 0; j < 8; ++j)
            acc[j] = WMMA_BF16(a.v, bfrag[j], acc[j]);
    }

    // C/D layout: VGPR i -> row (i + 8*hi), col = lo. Fuse bias, store (S,B,H).
#pragma unroll
    for (int tl = 0; tl < 4; ++tl) {
        const int o = c0 + tl * 16 + lo;
        const float biasR = wr_bias[o];
        const float biasW = w_bias[o];
#pragma unroll
        for (int i = 0; i < 8; ++i) {
            int rIdx = rowBase + hi * 8 + i;    // flat (b,s)
            int b = rIdx >> 7;                  // / S_DIM
            int s = rIdx & (S_DIM - 1);
            size_t idx = ((size_t)s * B_DIM + b) * H_DIM + (size_t)o;
            fwr_out[idx] = acc[tl][i] + biasR;
            fw_out[idx]  = acc[4 + tl][i] + biasW;
        }
    }
}

// ---------------------------------------------------------------------------
// Phase 2: persistent GRU scan. The recurrence is independent per batch row,
// so each block owns 16 rows of h and runs all 128 steps with only a
// workgroup barrier per step (no kernel launches, no global h traffic).
//  - h (bf16) lives in double-buffered LDS for cross-wave WMMA A fragments.
//  - h (f32) lives in registers: each lane owns the same 16 output cells
//    every step (C-layout is fixed), so the precise copy never moves.
//  - K-loop is software-pipelined (ping-pong fragment buffers) so each
//    4-WMMA chain covers the L2 latency of the next chunk's loads.
//  - Elementwise-tail operands (fWr/fW/G) are loaded BEFORE the GEMM so the
//    whole K-loop hides their HBM latency; next step's slice is prefetched.
// 16 waves/block, each wave: 2 col-tiles x 2 matrices = 4 WMMA chains.
// ---------------------------------------------------------------------------
__global__ __launch_bounds__(512) void gru_scan(
    const float* __restrict__ fwr,     // (S,B,H), bias fused
    const float* __restrict__ fw,      // (S,B,H), bias fused
    const float* __restrict__ G,       // (B,S)
    const bf16*  __restrict__ ur16, const bf16* __restrict__ u16,
    const float* __restrict__ ur_bias, const float* __restrict__ u_bias,
    float* __restrict__ out)           // (B,H)
{
    __shared__ bf16 hB[2][16 * LDSB_STRIDE];

    const int lane = threadIdx.x & 31;
    const int wave = threadIdx.x >> 5;      // 0..15
    const int hi = lane >> 4;
    const int lo = lane & 15;
    const int rowBase = blockIdx.x * 16;    // batch rows owned by this block
    const int c0 = wave * 32;               // columns [c0, c0+32)

    // h0 = 0
    for (int i = threadIdx.x; i < 16 * LDSB_STRIDE; i += 512) hB[0][i] = (bf16)0.0f;
    float hp0[8], hp1[8];                   // f32 h for this lane's cells
#pragma unroll
    for (int i = 0; i < 8; ++i) { hp0[i] = 0.0f; hp1[i] = 0.0f; }
    __syncthreads();

    const int o0 = c0 + lo;
    const int o1 = c0 + 16 + lo;
    const float urb0 = ur_bias[o0], urb1 = ur_bias[o1];
    const float ub0  = u_bias[o0],  ub1  = u_bias[o1];
    const bf16* urP0 = ur16 + (size_t)o0 * H_DIM;
    const bf16* urP1 = ur16 + (size_t)o1 * H_DIM;
    const bf16* uP0  = u16  + (size_t)o0 * H_DIM;
    const bf16* uP1  = u16  + (size_t)o1 * H_DIM;

    for (int t = 0; t < S_DIM; ++t) {
        const int p = t & 1, q = p ^ 1;
        const float* fwr_t = fwr + ((size_t)t * B_DIM + rowBase) * H_DIM;
        const float* fw_t  = fw  + ((size_t)t * B_DIM + rowBase) * H_DIM;

        // Prefetch next step's 2 x 32KB slice: 512 threads x 64B covers it.
        if (t + 1 < S_DIM) {
            __builtin_prefetch((const char*)(fwr_t + (size_t)B_DIM * H_DIM) + threadIdx.x * 64, 0, 3);
            __builtin_prefetch((const char*)(fw_t  + (size_t)B_DIM * H_DIM) + threadIdx.x * 64, 0, 3);
        }

        // Issue the tail's operand loads early: GEMM hides their latency.
        float fr0[8], fr1[8], fc0[8], fc1[8], gv[8];
#pragma unroll
        for (int i = 0; i < 8; ++i) {
            const int m = hi * 8 + i;
            const size_t fidx = (size_t)m * H_DIM;
            fr0[i] = fwr_t[fidx + o0];
            fr1[i] = fwr_t[fidx + o1];
            fc0[i] = fw_t[fidx + o0];
            fc1[i] = fw_t[fidx + o1];
            gv[i]  = G[(size_t)(rowBase + m) * S_DIM + t];
        }

        v8f aR0 = {}, aR1 = {}, aU0 = {}, aU1 = {};

        // Software-pipelined K loop: ping-pong fragment sets A/B.
        union Af { v16bf v; v8bf h[2]; };
        const bf16* hRow = &hB[p][lo * LDSB_STRIDE + hi * 16];

        Af aA, aB;
        v16bf r0A, r1A, u0A, u1A, r0B, r1B, u0B, u1B;

        // prologue: set A at kc = 0
        aA.h[0] = *(const v8bf*)(hRow);
        aA.h[1] = *(const v8bf*)(hRow + 8);
        r0A = *(const v16bf*)(urP0 + hi * 16);
        r1A = *(const v16bf*)(urP1 + hi * 16);
        u0A = *(const v16bf*)(uP0  + hi * 16);
        u1A = *(const v16bf*)(uP1  + hi * 16);

#pragma unroll
        for (int kc = 0; kc < H_DIM; kc += 64) {
            // load set B at kc+32 (always in range: kc <= 448)
            {
                const int k1 = kc + 32;
                aB.h[0] = *(const v8bf*)(hRow + k1);
                aB.h[1] = *(const v8bf*)(hRow + k1 + 8);
                r0B = *(const v16bf*)(urP0 + k1 + hi * 16);
                r1B = *(const v16bf*)(urP1 + k1 + hi * 16);
                u0B = *(const v16bf*)(uP0  + k1 + hi * 16);
                u1B = *(const v16bf*)(uP1  + k1 + hi * 16);
            }
            // compute on set A (kc)
            aR0 = WMMA_BF16(aA.v, r0A, aR0);
            aR1 = WMMA_BF16(aA.v, r1A, aR1);
            aU0 = WMMA_BF16(aA.v, u0A, aU0);
            aU1 = WMMA_BF16(aA.v, u1A, aU1);

            // load set A at kc+64 (guarded; static after unroll)
            if (kc + 64 < H_DIM) {
                const int k2 = kc + 64;
                aA.h[0] = *(const v8bf*)(hRow + k2);
                aA.h[1] = *(const v8bf*)(hRow + k2 + 8);
                r0A = *(const v16bf*)(urP0 + k2 + hi * 16);
                r1A = *(const v16bf*)(urP1 + k2 + hi * 16);
                u0A = *(const v16bf*)(uP0  + k2 + hi * 16);
                u1A = *(const v16bf*)(uP1  + k2 + hi * 16);
            }
            // compute on set B (kc+32)
            aR0 = WMMA_BF16(aB.v, r0B, aR0);
            aR1 = WMMA_BF16(aB.v, r1B, aR1);
            aU0 = WMMA_BF16(aB.v, u0B, aU0);
            aU1 = WMMA_BF16(aB.v, u1B, aU1);
        }

        // Elementwise tail: gate, blend, publish h.
#pragma unroll
        for (int i = 0; i < 8; ++i) {
            const int m = hi * 8 + i;
            const int b = rowBase + m;
            const float g = gv[i];

            {   // column tile 0
                float r    = 1.0f / (1.0f + __expf(-(fr0[i] + aR0[i] + urb0)));
                float cand = tanhf(fc0[i] + r * (aU0[i] + ub0));
                float hn   = g * cand + (1.0f - g) * hp0[i];
                hp0[i] = hn;
                hB[q][m * LDSB_STRIDE + o0] = (bf16)hn;
                if (t == S_DIM - 1) out[(size_t)b * H_DIM + o0] = hn;
            }
            {   // column tile 1
                float r    = 1.0f / (1.0f + __expf(-(fr1[i] + aR1[i] + urb1)));
                float cand = tanhf(fc1[i] + r * (aU1[i] + ub1));
                float hn   = g * cand + (1.0f - g) * hp1[i];
                hp1[i] = hn;
                hB[q][m * LDSB_STRIDE + o1] = (bf16)hn;
                if (t == S_DIM - 1) out[(size_t)b * H_DIM + o1] = hn;
            }
        }
        __syncthreads();   // publish hB[q] before next step's fragment reads
    }
}

// ---------------------------------------------------------------------------
// Launch
// ---------------------------------------------------------------------------
extern "C" void kernel_launch(void* const* d_in, const int* in_sizes, int n_in,
                              void* d_out, int out_size, void* d_ws, size_t ws_size,
                              hipStream_t stream) {
    (void)in_sizes; (void)n_in; (void)out_size; (void)ws_size;

    const float* facts = (const float*)d_in[0];
    const float* G     = (const float*)d_in[1];
    const float* Wr_w  = (const float*)d_in[2];
    const float* Wr_b  = (const float*)d_in[3];
    const float* Ur_w  = (const float*)d_in[4];
    const float* Ur_b  = (const float*)d_in[5];
    const float* W_w   = (const float*)d_in[6];
    const float* W_b   = (const float*)d_in[7];
    const float* U_w   = (const float*)d_in[8];
    const float* U_b   = (const float*)d_in[9];

    const size_t SBH = (size_t)S_DIM * B_DIM * H_DIM;
    const size_t HH  = (size_t)H_DIM * H_DIM;

    char* ws = (char*)d_ws;
    float* fwr  = (float*)ws; ws += SBH * sizeof(float);
    float* fw   = (float*)ws; ws += SBH * sizeof(float);
    bf16*  wr16 = (bf16*)ws; ws += HH * sizeof(bf16);
    bf16*  ur16 = (bf16*)ws; ws += HH * sizeof(bf16);
    bf16*  w16  = (bf16*)ws; ws += HH * sizeof(bf16);
    bf16*  u16  = (bf16*)ws; ws += HH * sizeof(bf16);

    cvt_weights<<<(int)((HH + 255) / 256), 256, 0, stream>>>(
        Wr_w, Ur_w, W_w, U_w, wr16, ur16, w16, u16);

    // Phase 1: 4096 row-stripes; 256 threads (8 waves) per block.
    fact_gemm<<<4096, 256, 0, stream>>>(facts, wr16, w16, Wr_b, W_b, fwr, fw);

    // Phase 2: persistent scan, 32 blocks x 512 threads, zero re-launches.
    gru_scan<<<32, 512, 0, stream>>>(fwr, fw, G, ur16, u16, Ur_b, U_b, (float*)d_out);
}
